// MambaCrossBlock_30623116820624
// MI455X (gfx1250) — matrix operand order repaired and checked by reading, not compiled, not run on hardware
//
#include <hip/hip_runtime.h>
#include <hip/hip_bf16.h>
#include <math.h>

#define EPS 1e-5f
#define DM 512
#define DS 16
#define DI 1024
#define INCH 256
#define SEQL 1024
#define NIMG 4

typedef __bf16 bf16;
typedef bf16 bf16x16 __attribute__((ext_vector_type(16)));
typedef float f32x8 __attribute__((ext_vector_type(8)));

enum { EPI_PLAIN = 0, EPI_BNRELU = 1, EPI_BNRES = 2 };

// ---------------------------------------------------------------------------
// Generic bf16-WMMA GEMM: D[img] = W * X[img]  (W: MxK row-major, X: KxN)
// 4 waves / block, 64x64 tile, K-step 32, f32 accumulate.
// LNX: apply per-column LayerNorm to X while loading the B tile.
// EPI: fused epilogue (BN+ReLU for cr-conv, BN+residual for rs-conv).
// ---------------------------------------------------------------------------
template<int EPI, bool LNX>
__global__ __launch_bounds__(128)
void gemm_wmma_kernel(const float* __restrict__ W0, const float* __restrict__ W1,
                      const float* __restrict__ X0, const float* __restrict__ X1,
                      long xImgStride,
                      float* __restrict__ D,
                      int M, int N, int K,
                      const float* __restrict__ bnG, const float* __restrict__ bnB,
                      const float* __restrict__ bnM, const float* __restrict__ bnV,
                      const float* __restrict__ lnG0, const float* __restrict__ lnB0,
                      const float* __restrict__ lnG1, const float* __restrict__ lnB1,
                      const float* __restrict__ colMu, const float* __restrict__ colRinv,
                      const float* __restrict__ R0, const float* __restrict__ R1,
                      long rImgStride)
{
    __shared__ bf16 sA[64][40];   // [m][k], padded stride
    __shared__ bf16 sB[64][40];   // [n][k], padded stride

    const int img   = blockIdx.z;
    const int mod   = img >> 1;
    const int nBase = blockIdx.x * 64;
    const int mBase = blockIdx.y * 64;
    const int tid   = threadIdx.x;
    const int lane  = tid & 31;
    const int wave  = tid >> 5;
    const int wr    = wave >> 1, wc = wave & 1;
    const int half  = lane >> 4;
    const int l16   = lane & 15;

    const float* W   = mod ? W1 : W0;
    const float* X   = (mod ? X1 : X0) + (long)(img & 1) * xImgStride;
    const float* lnG = mod ? lnG1 : lnG0;
    const float* lnB = mod ? lnB1 : lnB0;
    float* Dimg = D + (long)img * M * N;

    f32x8 acc[2][2] = {};

    for (int k0 = 0; k0 < K; k0 += 32) {
        // --- A tile: 64x32 fp32 -> bf16, [m][k] ---
        #pragma unroll
        for (int j = 0; j < 4; ++j) {
            int idx = tid + 128 * j;
            int row = idx >> 3;
            int c4  = (idx & 7) * 4;
            float4 w4 = *(const float4*)(W + (long)(mBase + row) * K + k0 + c4);
            sA[row][c4 + 0] = (bf16)w4.x;
            sA[row][c4 + 1] = (bf16)w4.y;
            sA[row][c4 + 2] = (bf16)w4.z;
            sA[row][c4 + 3] = (bf16)w4.w;
        }
        // --- B tile: 32x64 fp32, transposed into [n][k], optional fused LN ---
        #pragma unroll
        for (int j = 0; j < 4; ++j) {
            int idx = tid + 128 * j;
            int k   = idx >> 4;
            int n4  = (idx & 15) * 4;
            float4 x4 = *(const float4*)(X + (long)(k0 + k) * N + nBase + n4);
            float vals[4] = {x4.x, x4.y, x4.z, x4.w};
            #pragma unroll
            for (int e = 0; e < 4; ++e) {
                float v = vals[e];
                if (LNX) {
                    int n = nBase + n4 + e;
                    v = (v - colMu[img * N + n]) * colRinv[img * N + n]
                        * lnG[k0 + k] + lnB[k0 + k];
                }
                sB[n4 + e][k] = (bf16)v;
            }
        }
        __syncthreads();

        // --- fragment loads per ISA 16-bit A(16x32)/B(32x16) lane layout ---
        union U { bf16x16 v; uint4 q[2]; };
        U a0, a1, b0, b1;
        int m0  = wr * 32 + l16;
        int nn0 = wc * 32 + l16;
        a0.q[0] = *(const uint4*)&sA[m0][half * 8];
        a0.q[1] = *(const uint4*)&sA[m0][16 + half * 8];
        a1.q[0] = *(const uint4*)&sA[m0 + 16][half * 8];
        a1.q[1] = *(const uint4*)&sA[m0 + 16][16 + half * 8];
        b0.q[0] = *(const uint4*)&sB[nn0][half * 8];
        b0.q[1] = *(const uint4*)&sB[nn0][16 + half * 8];
        b1.q[0] = *(const uint4*)&sB[nn0 + 16][half * 8];
        b1.q[1] = *(const uint4*)&sB[nn0 + 16][16 + half * 8];

        acc[0][0] = __builtin_amdgcn_wmma_f32_16x16x32_bf16(false, a0.v, false, b0.v, (short)0, acc[0][0], false, false);
        acc[0][1] = __builtin_amdgcn_wmma_f32_16x16x32_bf16(false, a0.v, false, b1.v, (short)0, acc[0][1], false, false);
        acc[1][0] = __builtin_amdgcn_wmma_f32_16x16x32_bf16(false, a1.v, false, b0.v, (short)0, acc[1][0], false, false);
        acc[1][1] = __builtin_amdgcn_wmma_f32_16x16x32_bf16(false, a1.v, false, b1.v, (short)0, acc[1][1], false, false);

        __syncthreads();
    }

    // --- fused epilogue + store (C/D layout: elem r -> row r + 8*half) ---
    const float* Rr = nullptr;
    if (EPI == EPI_BNRES) Rr = (mod ? R1 : R0) + (long)(img & 1) * rImgStride;
    #pragma unroll
    for (int ti = 0; ti < 2; ++ti) {
        #pragma unroll
        for (int tj = 0; tj < 2; ++tj) {
            int n = nBase + wc * 32 + tj * 16 + l16;
            #pragma unroll
            for (int r = 0; r < 8; ++r) {
                int m = mBase + wr * 32 + ti * 16 + r + 8 * half;
                float v = acc[ti][tj][r];
                if (EPI == EPI_BNRELU) {
                    v = (v - bnM[m]) * rsqrtf(bnV[m] + EPS) * bnG[m] + bnB[m];
                    v = fmaxf(v, 0.f);
                } else if (EPI == EPI_BNRES) {
                    v = (v - bnM[m]) * rsqrtf(bnV[m] + EPS) * bnG[m] + bnB[m]
                        + Rr[(long)m * N + n];
                }
                Dimg[(long)m * N + n] = v;
            }
        }
    }
}

// ---------------------------------------------------------------------------
// Per-column LayerNorm stats over D_MODEL for each (img, position)
// ---------------------------------------------------------------------------
__global__ void ln_stats_kernel(const float* __restrict__ xr,
                                float* __restrict__ mu, float* __restrict__ rinv)
{
    int gid = blockIdx.x * blockDim.x + threadIdx.x;   // 4096
    int p = gid & (SEQL - 1);
    int img = gid >> 10;
    const float* base = xr + (long)img * DM * SEQL + p;
    float s = 0.f, ss = 0.f;
    for (int d = 0; d < DM; ++d) {
        float v = base[(long)d * SEQL];
        s += v; ss += v * v;
    }
    float m = s * (1.f / DM);
    float var = ss * (1.f / DM) - m * m;
    mu[gid]   = m;
    rinv[gid] = rsqrtf(var + EPS);
}

// ---------------------------------------------------------------------------
// Lambda head: global-avg-pool -> 128 ReLU -> sigmoid, one block per img
// ---------------------------------------------------------------------------
__global__ __launch_bounds__(256)
void lambda_kernel(const float* __restrict__ xr,
                   const float* __restrict__ w1V, const float* __restrict__ b1V,
                   const float* __restrict__ w2V, const float* __restrict__ b2V,
                   const float* __restrict__ w1I, const float* __restrict__ b1I,
                   const float* __restrict__ w2I, const float* __restrict__ b2I,
                   float* __restrict__ lamOut)
{
    __shared__ float sp[DM];
    __shared__ float sh[128];
    int img = blockIdx.x, t = threadIdx.x;
    const float* xi = xr + (long)img * DM * SEQL;
    for (int d = t; d < DM; d += 256) {
        float s = 0.f;
        const float* row = xi + (long)d * SEQL;
        for (int p = 0; p < SEQL; ++p) s += row[p];
        sp[d] = s * (1.f / SEQL);
    }
    __syncthreads();
    const float* w1 = (img < 2) ? w1V : w1I;
    const float* b1 = (img < 2) ? b1V : b1I;
    if (t < 128) {
        float a = b1[t];
        for (int k = 0; k < DM; ++k) a += w1[(long)t * DM + k] * sp[k];
        sh[t] = fmaxf(a, 0.f);
    }
    __syncthreads();
    if (t == 0) {
        const float* w2 = (img < 2) ? w2V : w2I;
        const float* b2 = (img < 2) ? b2V : b2I;
        float a = b2[0];
        for (int k = 0; k < 128; ++k) a += w2[k] * sh[k];
        lamOut[img] = 1.f / (1.f + __expf(-a));
    }
}

// ---------------------------------------------------------------------------
// SSM scans: one thread per (bsel, img, channel). fwd writes y; bwd read-adds.
// ---------------------------------------------------------------------------
__global__ __launch_bounds__(256)
void ssm_scan_kernel(const float* __restrict__ xp,
                     const float* __restrict__ AlogV, const float* __restrict__ AlogI,
                     const float* __restrict__ BV, const float* __restrict__ BI,
                     const float* __restrict__ CV, const float* __restrict__ CI,
                     float* __restrict__ acc)
{
    int gid  = blockIdx.x * blockDim.x + threadIdx.x;   // 8192
    int i    = gid & (DI - 1);
    int img  = (gid >> 10) & 3;
    int bsel = gid >> 12;                                // 0=std, 1=cross
    int mod  = img >> 1;
    const float* Alog = mod ? AlogI : AlogV;
    const float* Cm   = mod ? CI : CV;
    int bmod = bsel ? (1 - mod) : mod;
    const float* Bm   = bmod ? BI : BV;

    float Av[DS], Bv[DS], Cv[DS], h[DS];
    #pragma unroll
    for (int s = 0; s < DS; ++s) {
        Av[s] = -__expf(Alog[i * DS + s]);
        Bv[s] = Bm[i * DS + s];
        Cv[s] = Cm[i * DS + s];
        h[s]  = 0.f;
    }
    const float* x = xp + (long)img * (2 * DI) * SEQL + (long)i * SEQL;
    float* y = acc + ((long)bsel * NIMG + img) * DI * SEQL + (long)i * SEQL;

    for (int t0 = 0; t0 < SEQL; t0 += 4) {
        float4 x4 = *(const float4*)(x + t0);
        __builtin_prefetch(x + t0 + 64, 0, 0);   // global_prefetch_b8
        float xv[4] = {x4.x, x4.y, x4.z, x4.w};
        float yv[4];
        #pragma unroll
        for (int e = 0; e < 4; ++e) {
            float xt = xv[e], d = 0.f;
            #pragma unroll
            for (int s = 0; s < DS; ++s) {
                h[s] = __builtin_fmaf(h[s], Av[s], xt * Bv[s]);
                d = __builtin_fmaf(h[s], Cv[s], d);
            }
            yv[e] = d;
        }
        *(float4*)(y + t0) = make_float4(yv[0], yv[1], yv[2], yv[3]);
    }
    #pragma unroll
    for (int s = 0; s < DS; ++s) h[s] = 0.f;
    for (int t0 = SEQL - 4; t0 >= 0; t0 -= 4) {
        float4 x4 = *(const float4*)(x + t0);
        float4 yf = *(const float4*)(y + t0);
        float xv[4] = {x4.x, x4.y, x4.z, x4.w};
        float yv[4] = {yf.x, yf.y, yf.z, yf.w};
        #pragma unroll
        for (int e = 3; e >= 0; --e) {
            float xt = xv[e], d = 0.f;
            #pragma unroll
            for (int s = 0; s < DS; ++s) {
                h[s] = __builtin_fmaf(h[s], Av[s], xt * Bv[s]);
                d = __builtin_fmaf(h[s], Cv[s], d);
            }
            yv[e] += d;
        }
        *(float4*)(y + t0) = make_float4(yv[0], yv[1], yv[2], yv[3]);
    }
}

// ---------------------------------------------------------------------------
// comb = 0.5*silu(gate)*(lam*(fwd+bwd)_std + (1-lam)*(fwd+bwd)_cross)
// ---------------------------------------------------------------------------
__global__ void combine_kernel(const float* __restrict__ xp,
                               const float* __restrict__ acc,
                               const float* __restrict__ lam,
                               float* __restrict__ comb)
{
    long gid = (long)blockIdx.x * blockDim.x + threadIdx.x;   // 4M
    if (gid >= (long)NIMG * DI * SEQL) return;
    int t   = (int)(gid & (SEQL - 1));
    int i   = (int)((gid >> 10) & (DI - 1));
    int img = (int)(gid >> 20);
    float g = xp[(long)img * (2 * DI) * SEQL + (long)(DI + i) * SEQL + t];
    float silu = g / (1.f + __expf(-g));
    float la = lam[img];
    long idx = (long)img * DI * SEQL + (long)i * SEQL + t;
    float sS = acc[idx];
    float sX = acc[(long)NIMG * DI * SEQL + idx];
    comb[idx] = 0.5f * silu * (la * sS + (1.f - la) * sX);
}

// ---------------------------------------------------------------------------
extern "C" void kernel_launch(void* const* d_in, const int* in_sizes, int n_in,
                              void* d_out, int out_size, void* d_ws, size_t ws_size,
                              hipStream_t stream) {
    const float* cr_w = (const float*)d_in[0];
    const float* cr_g = (const float*)d_in[1];
    const float* cr_b = (const float*)d_in[2];
    const float* cr_m = (const float*)d_in[3];
    const float* cr_v = (const float*)d_in[4];
    const float* rs_w = (const float*)d_in[5];
    const float* rs_g = (const float*)d_in[6];
    const float* rs_b = (const float*)d_in[7];
    const float* rs_m = (const float*)d_in[8];
    const float* rs_v = (const float*)d_in[9];
    const float* mV_ln_g = (const float*)d_in[10];
    const float* mV_ln_b = (const float*)d_in[11];
    const float* mV_in_w = (const float*)d_in[12];
    const float* mV_A    = (const float*)d_in[13];
    const float* mV_B    = (const float*)d_in[14];
    const float* mV_C    = (const float*)d_in[15];
    const float* mV_out_w= (const float*)d_in[16];
    const float* lpV_w1  = (const float*)d_in[17];
    const float* lpV_b1  = (const float*)d_in[18];
    const float* lpV_w2  = (const float*)d_in[19];
    const float* lpV_b2  = (const float*)d_in[20];
    const float* mI_ln_g = (const float*)d_in[21];
    const float* mI_ln_b = (const float*)d_in[22];
    const float* mI_in_w = (const float*)d_in[23];
    const float* mI_A    = (const float*)d_in[24];
    const float* mI_B    = (const float*)d_in[25];
    const float* mI_C    = (const float*)d_in[26];
    const float* mI_out_w= (const float*)d_in[27];
    const float* lpI_w1  = (const float*)d_in[28];
    const float* lpI_b1  = (const float*)d_in[29];
    const float* lpI_w2  = (const float*)d_in[30];
    const float* lpI_b2  = (const float*)d_in[31];
    const float* x_V = (const float*)d_in[32];
    const float* x_I = (const float*)d_in[33];

    // workspace layout (floats); all intermediates fit in 192MB L2
    float* ws   = (float*)d_ws;
    float* xr   = ws;                             // 4*512*1024
    float* xp   = xr   + (long)NIMG*DM*SEQL;      // 4*2048*1024
    float* acc  = xp   + (long)NIMG*2*DI*SEQL;    // 2*4*1024*1024
    float* comb = acc  + (long)2*NIMG*DI*SEQL;    // 4*1024*1024
    float* mout = comb + (long)NIMG*DI*SEQL;      // 4*512*1024
    float* mu   = mout + (long)NIMG*DM*SEQL;      // 4096
    float* rinv = mu   + NIMG*SEQL;               // 4096
    float* lam  = rinv + NIMG*SEQL;               // 4

    dim3 blk(128);

    // 1) cr-conv + BN + ReLU  -> xr[img][512][1024]
    gemm_wmma_kernel<EPI_BNRELU, false><<<dim3(SEQL/64, DM/64, NIMG), blk, 0, stream>>>(
        cr_w, cr_w, x_V, x_I, (long)INCH*SEQL, xr, DM, SEQL, INCH,
        cr_g, cr_b, cr_m, cr_v,
        nullptr, nullptr, nullptr, nullptr, nullptr, nullptr,
        nullptr, nullptr, 0);

    // 2) LN column stats + lambda head
    ln_stats_kernel<<<dim3(NIMG*SEQL/256), dim3(256), 0, stream>>>(xr, mu, rinv);
    lambda_kernel<<<dim3(NIMG), dim3(256), 0, stream>>>(
        xr, lpV_w1, lpV_b1, lpV_w2, lpV_b2, lpI_w1, lpI_b1, lpI_w2, lpI_b2, lam);

    // 3) in-proj with fused LN -> xp[img][2048][1024] (x_ssm rows 0..1023, gate 1024..2047)
    gemm_wmma_kernel<EPI_PLAIN, true><<<dim3(SEQL/64, (2*DI)/64, NIMG), blk, 0, stream>>>(
        mV_in_w, mI_in_w, xr, xr + (long)2*DM*SEQL, (long)DM*SEQL, xp, 2*DI, SEQL, DM,
        nullptr, nullptr, nullptr, nullptr,
        mV_ln_g, mV_ln_b, mI_ln_g, mI_ln_b, mu, rinv,
        nullptr, nullptr, 0);

    // 4) 16 fwd/bwd SSM scans (std + cross B), fwd+bwd summed per thread
    ssm_scan_kernel<<<dim3(2*NIMG*DI/256), dim3(256), 0, stream>>>(
        xp, mV_A, mI_A, mV_B, mI_B, mV_C, mI_C, acc);

    // 5) silu(gate) * lambda blend (before linear out-proj, since it commutes)
    combine_kernel<<<dim3((NIMG*DI*SEQL)/256), dim3(256), 0, stream>>>(xp, acc, lam, comb);

    // 6) out-proj -> mout[img][512][1024]
    gemm_wmma_kernel<EPI_PLAIN, false><<<dim3(SEQL/64, DM/64, NIMG), blk, 0, stream>>>(
        mV_out_w, mI_out_w, comb, comb + (long)2*DI*SEQL, (long)DI*SEQL, mout, DM, SEQL, DI,
        nullptr, nullptr, nullptr, nullptr,
        nullptr, nullptr, nullptr, nullptr,
        nullptr, nullptr,
        nullptr, nullptr, 0);

    // 7) rs-conv + BN + residual -> d_out (V imgs then I imgs, contiguous)
    gemm_wmma_kernel<EPI_BNRES, false><<<dim3(SEQL/64, INCH/64, NIMG), blk, 0, stream>>>(
        rs_w, rs_w, mout, mout + (long)2*DM*SEQL, (long)DM*SEQL, (float*)d_out, INCH, SEQL, DM,
        rs_g, rs_b, rs_m, rs_v,
        nullptr, nullptr, nullptr, nullptr, nullptr, nullptr,
        x_V, x_I, (long)INCH*SEQL);

    (void)in_sizes; (void)n_in; (void)out_size; (void)ws_size;
}
